// CKANKANNet_80736795231207
// MI455X (gfx1250) — matrix operand order repaired and checked by reading, not compiled, not run on hardware
//
#include <hip/hip_runtime.h>
#include <hip/hip_bf16.h>

// ---------- CDNA5 WMMA types ----------
typedef __bf16 v16bf __attribute__((ext_vector_type(16)));
typedef float  v8f   __attribute__((ext_vector_type(8)));

union Frag {
    v16bf v;
    unsigned int d[8];
    unsigned short u[16];
};

__device__ __forceinline__ unsigned short f2bf(float f) {
    union { float f; unsigned int u; } cv; cv.f = f;
    unsigned int r = cv.u + 0x7FFFu + ((cv.u >> 16) & 1u);   // round-to-nearest-even
    return (unsigned short)(r >> 16);
}

// ---------- B-spline expansion: x -> [silu(x), B_0..B_7(x)] ----------
// GRID=5, DEGREE=3, range [-1,1] -> h=0.4, knots pts[i] = -2.2 + 0.4*i (12 knots)
__device__ __forceinline__ void kan_features(float x, float out[9]) {
    float sig = 1.0f / (1.0f + __expf(-x));
    out[0] = x * sig;                                  // silu
    float bb[11];
#pragma unroll
    for (int j = 0; j < 11; ++j) {
        float p0 = -2.2f + 0.4f * j;
        bb[j] = (x >= p0 && x < p0 + 0.4f) ? 1.0f : 0.0f;
    }
#pragma unroll
    for (int d = 1; d <= 3; ++d) {
        float inv = 1.0f / (0.4f * (float)d);          // uniform knots: both denoms = d*h
#pragma unroll 10
        for (int j = 0; j <= 10 - d; ++j) {
            float pj   = -2.2f + 0.4f * j;
            float pjd1 = -2.2f + 0.4f * (j + d + 1);
            bb[j] = (x - pj) * inv * bb[j] + (pjd1 - x) * inv * bb[j + 1];
        }
    }
#pragma unroll
    for (int j = 0; j < 8; ++j) out[1 + j] = bb[j];
}

// Expand NCHW fp32 -> [B][C*9][H][W] bf16 (channel ce = c*9 + j)
__global__ void kan_expand_nchw(const float* __restrict__ X, unsigned short* __restrict__ E,
                                int HW, int total) {
    int idx = blockIdx.x * blockDim.x + threadIdx.x;
    if (idx >= total) return;
    int pos = idx % HW;
    int bc  = idx / HW;                   // b*C + c
    float out[9];
    kan_features(X[idx], out);
    size_t base = ((size_t)bc * 9) * HW + pos;
#pragma unroll
    for (int j = 0; j < 9; ++j) E[base + (size_t)j * HW] = f2bf(out[j]);
}

// Expand flat features -> row-major [B][F*9] bf16 (k = f*9 + j)
__global__ void kan_expand_rowmajor(const float* __restrict__ X, unsigned short* __restrict__ E,
                                    int total) {
    int idx = blockIdx.x * blockDim.x + threadIdx.x;
    if (idx >= total) return;
    float out[9];
    kan_features(X[idx], out);
    size_t base = (size_t)idx * 9;
#pragma unroll
    for (int j = 0; j < 9; ++j) E[base + j] = f2bf(out[j]);
}

// ---------- weight packing ----------
// Combined conv weight -> Wp[k][n], k = (c*9 + j)*9 + (dy*3+dx), K = C*81
__global__ void pack_conv_w(const float* __restrict__ wb, const float* __restrict__ wsp,
                            unsigned short* __restrict__ Wp, int C, int O, int Ktotal) {
    int idx = blockIdx.x * blockDim.x + threadIdx.x;
    if (idx >= Ktotal * O) return;
    int k = idx / O, n = idx % O;
    int ce = k / 9, r = k % 9;
    int c = ce / 9, j = ce % 9;
    float v = (j == 0) ? wb[((size_t)n * C + c) * 9 + r]
                       : wsp[((size_t)n * C * 8 + (size_t)c * 8 + (j - 1)) * 9 + r];
    Wp[idx] = f2bf(v);
}

// Linear weight -> row-major [112][36864] bf16 (rows 100..111 zero-padded)
__global__ void pack_linear_w(const float* __restrict__ lb, const float* __restrict__ lc,
                              unsigned short* __restrict__ Wp, int total) {
    int idx = blockIdx.x * blockDim.x + threadIdx.x;
    if (idx >= total) return;
    int n = idx / 36864, k = idx % 36864;
    int f = k / 9, j = k % 9;
    float v = 0.0f;
    if (n < 100) v = (j == 0) ? lb[(size_t)n * 4096 + f]
                              : lc[((size_t)n * 4096 + f) * 8 + (j - 1)];
    Wp[idx] = f2bf(v);
}

__global__ void zero_f32(float* __restrict__ p, int n) {
    int i = blockIdx.x * blockDim.x + threadIdx.x;
    if (i < n) p[i] = 0.0f;
}

__global__ void maxpool2x2(const float* __restrict__ In, float* __restrict__ Out,
                           int H, int W, int total) {
    int idx = blockIdx.x * blockDim.x + threadIdx.x;
    if (idx >= total) return;
    int W2 = W >> 1, H2 = H >> 1;
    int x = idx % W2; int t = idx / W2; int y = t % H2; int bc = t / H2;
    const float* p = In + ((size_t)bc * H + 2 * y) * W + 2 * x;
    Out[idx] = fmaxf(fmaxf(p[0], p[1]), fmaxf(p[W], p[W + 1]));
}

// ---------- implicit-GEMM conv, bf16 WMMA, f32 accumulate ----------
// Out[m,n] = sum_k E[b, ce, h+dy-1, w+dx-1] * Wp[k][n], m->(b,h,w), k->(ce,dy,dx)
// Block: 256 thr (8 waves). Tile: 128(M) x 64(N), K staged 64/iter via LDS (8 WMMA/wave/stage).
__global__ __launch_bounds__(256)
void kan_conv_wmma(const unsigned short* __restrict__ E,   // [B][C9][H][W] bf16
                   const unsigned short* __restrict__ Wp,  // [K][O] bf16
                   float* __restrict__ Out,                // [B][O][H][W] f32
                   int C9, int H, int W, int O, int Mtotal, int Ktotal) {
    __shared__ __align__(16) unsigned short lds_a[128][68];  // 64 + 4 pad (bank spread)
    __shared__ __align__(16) unsigned short lds_b[64][68];   // stored [n][k]

    const int tid  = threadIdx.x;
    const int wave = tid >> 5;
    const int lane = tid & 31;
    const int m0 = blockIdx.x * 128;
    const int n0 = blockIdx.y * 64;
    const int HW = H * W;

    v8f acc[4] = {v8f{}, v8f{}, v8f{}, v8f{}};

    // A staging: thread owns half-row: 32 consecutive k of row tid/2
    const int a_row = tid >> 1;
    const int a_k0  = (tid & 1) * 32;
    const int am = m0 + a_row;
    const bool am_ok = (am < Mtotal);
    int ab = 0, ah = 0, aw = 0;
    if (am_ok) { ab = am / HW; int r = am - ab * HW; ah = r / W; aw = r - ah * W; }
    const size_t abBase = (size_t)ab * C9 * HW;
    // running (ce, r) decomposition of k = k0 + a_k0 (carried incrementally, no divides
    // inside the K loop; advance by 64 = 7*9 + 1 each iteration)
    int ce_run = a_k0 / 9;
    int r_run  = a_k0 - ce_run * 9;

    // B staging: thread owns 16 consecutive n of one k row
    const int b_kl = tid >> 2;
    const int b_n0 = (tid & 3) * 16;

    // fragment addressing (A and B both use contiguous k = 16*(lane/16) + s)
    const int fr     = wave * 16 + (lane & 15);
    const int khalf  = (lane >> 4) * 8;     // unused; A also indexes by bkbase (K+16)
    const int bkbase = (lane >> 4) * 16;    // B: lanes 16-31 take K+16
    const int bn     = lane & 15;

    for (int k0 = 0; k0 < Ktotal; k0 += 64) {
        if (k0 + 64 < Ktotal)
            __builtin_prefetch(&Wp[(size_t)(k0 + 64) * O + n0 + (tid & 63)], 0, 0);
        // ---- stage A tile [128 x 64] (implicit-GEMM gather, zero padding)
        {
            int ce = ce_run, r = r_run;
            int kg = k0 + a_k0;
#pragma unroll 8
            for (int s = 0; s < 32; ++s) {
                unsigned short v = 0;
                if (am_ok && kg < Ktotal) {
                    int dy = r / 3;
                    int dx = r - dy * 3;
                    int y = ah + dy - 1, x = aw + dx - 1;
                    if ((unsigned)y < (unsigned)H && (unsigned)x < (unsigned)W)
                        v = E[abBase + (size_t)ce * HW + y * W + x];
                }
                lds_a[a_row][a_k0 + s] = v;
                ++kg;
                if (++r == 9) { r = 0; ++ce; }
            }
            // advance running decomposition by 64 for next iteration
            r_run += 1; ce_run += 7;
            if (r_run >= 9) { r_run -= 9; ce_run += 1; }
        }
        // ---- stage B tile [64 x 64] -> transposed lds_b[n][k]
        {
            int k = k0 + b_kl;
            const unsigned short* src = Wp + (size_t)k * O + n0 + b_n0;
#pragma unroll
            for (int s = 0; s < 16; ++s)
                lds_b[b_n0 + s][b_kl] = (k < Ktotal) ? src[s] : (unsigned short)0;
        }
        __syncthreads();

        // ---- build fragments, 8 WMMAs per wave (16x64 strip, K=64)
        Frag a[2];
#pragma unroll
        for (int h = 0; h < 2; ++h)
#pragma unroll
            for (int i = 0; i < 8; ++i) {
                int koff = bkbase + 2 * i;
                a[h].d[i] = *(const unsigned int*)&lds_a[fr][h * 32 + koff];
            }
#pragma unroll
        for (int nt = 0; nt < 4; ++nt) {
#pragma unroll
            for (int h = 0; h < 2; ++h) {
                Frag b;
#pragma unroll
                for (int i = 0; i < 8; ++i)
                    b.d[i] = *(const unsigned int*)&lds_b[nt * 16 + bn][h * 32 + bkbase + 2 * i];
                acc[nt] = __builtin_amdgcn_wmma_f32_16x16x32_bf16(
                    false, a[h].v, false, b.v, (short)0, acc[nt], false, false);
            }
        }
        __syncthreads();
    }

    // ---- epilogue: C/D layout -> NCHW f32
#pragma unroll
    for (int nt = 0; nt < 4; ++nt) {
        int n = n0 + nt * 16 + (lane & 15);
#pragma unroll
        for (int v = 0; v < 8; ++v) {
            int m = m0 + wave * 16 + v + (lane >> 4) * 8;
            if (m < Mtotal && n < O) {
                int b_ = m / HW; int r = m - b_ * HW; int h = r / W; int w = r - h * W;
                Out[((size_t)(b_ * O + n) * H + h) * W + w] = acc[nt][v];
            }
        }
    }
}

// ---------- KAN linear head: [64 x 36864] x [36864 x 100], split-K WMMA ----------
__global__ __launch_bounds__(32)
void kan_linear_wmma(const unsigned short* __restrict__ E,   // [64][36864] bf16
                     const unsigned short* __restrict__ Wp,  // [112][36864] bf16
                     float* __restrict__ Out) {              // [64][100] f32 (pre-zeroed)
    const int lane = threadIdx.x;
    const int m0 = blockIdx.x * 16;
    const int n0 = blockIdx.y * 16;
    const int kb0 = blockIdx.z * 1152;      // grid.z == 32 K-splits
    v8f acc = v8f{};
    const int khalf  = (lane >> 4) * 8;
    const int bkbase = (lane >> 4) * 16;
    const unsigned short* __restrict__ ea = E  + (size_t)(m0 + (lane & 15)) * 36864;
    const unsigned short* __restrict__ wb = Wp + (size_t)(n0 + (lane & 15)) * 36864;

    for (int kb = kb0; kb < kb0 + 1152; kb += 32) {
        Frag a, b;
#pragma unroll
        for (int i = 0; i < 8; ++i) {
            int koff = bkbase + 2 * i;
            a.d[i] = *(const unsigned int*)&ea[kb + koff];
        }
#pragma unroll
        for (int i = 0; i < 8; ++i)
            b.d[i] = *(const unsigned int*)&wb[kb + bkbase + 2 * i];
        acc = __builtin_amdgcn_wmma_f32_16x16x32_bf16(
            false, a.v, false, b.v, (short)0, acc, false, false);
    }
#pragma unroll
    for (int v = 0; v < 8; ++v) {
        int m = m0 + v + (lane >> 4) * 8;
        int n = n0 + (lane & 15);
        if (n < 100) atomicAdd(&Out[m * 100 + n], acc[v]);
    }
}

// =====================================================================
extern "C" void kernel_launch(void* const* d_in, const int* in_sizes, int n_in,
                              void* d_out, int out_size, void* d_ws, size_t ws_size,
                              hipStream_t stream) {
    (void)in_sizes; (void)n_in; (void)out_size; (void)ws_size;
    const float* x   = (const float*)d_in[0];
    const float* wb1 = (const float*)d_in[1];
    const float* ws1 = (const float*)d_in[2];
    const float* wb2 = (const float*)d_in[3];
    const float* ws2 = (const float*)d_in[4];
    const float* wb3 = (const float*)d_in[5];
    const float* ws3 = (const float*)d_in[6];
    const float* lb  = (const float*)d_in[7];
    const float* lc  = (const float*)d_in[8];
    float* out = (float*)d_out;
    char*  ws  = (char*)d_ws;

    // two ping-pong activation regions + weight region (total ~182 MB)
    const size_t OX = 0;
    const size_t OY = 84ull << 20;
    const size_t OW = 168ull << 20;
    unsigned short* E1 = (unsigned short*)(ws + OX);
    float*          C1 = (float*)(ws + OY);
    float*          P1 = (float*)(ws + OX);
    unsigned short* E2 = (unsigned short*)(ws + OY);
    float*          C2 = (float*)(ws + OX);
    float*          P2 = (float*)(ws + OY);
    unsigned short* E3 = (unsigned short*)(ws + OX);
    float*          C3 = (float*)(ws + OY);
    float*          P3 = (float*)(ws + OX);
    unsigned short* EL = (unsigned short*)(ws + OY);
    unsigned short* W1p = (unsigned short*)(ws + OW);
    unsigned short* W2p = (unsigned short*)(ws + OW + (1ull << 20));
    unsigned short* W3p = (unsigned short*)(ws + OW + (3ull << 20));
    unsigned short* WLp = (unsigned short*)(ws + OW + (5ull << 20));

    const int T = 256;
    auto nb = [](int n, int t) { return (n + t - 1) / t; };

    // ---- Layer 1: C=3 -> 64, 64x64 spatial
    kan_expand_nchw<<<nb(786432, T), T, 0, stream>>>(x, E1, 64 * 64, 786432);
    pack_conv_w<<<nb(243 * 64, T), T, 0, stream>>>(wb1, ws1, W1p, 3, 64, 243);
    kan_conv_wmma<<<dim3(262144 / 128, 1), T, 0, stream>>>(E1, W1p, C1, 27, 64, 64, 64, 262144, 243);
    maxpool2x2<<<nb(4194304, T), T, 0, stream>>>(C1, P1, 64, 64, 4194304);

    // ---- Layer 2: C=64 -> 128, 32x32 spatial
    kan_expand_nchw<<<nb(4194304, T), T, 0, stream>>>(P1, E2, 32 * 32, 4194304);
    pack_conv_w<<<nb(5184 * 128, T), T, 0, stream>>>(wb2, ws2, W2p, 64, 128, 5184);
    kan_conv_wmma<<<dim3(65536 / 128, 2), T, 0, stream>>>(E2, W2p, C2, 576, 32, 32, 128, 65536, 5184);
    maxpool2x2<<<nb(2097152, T), T, 0, stream>>>(C2, P2, 32, 32, 2097152);

    // ---- Layer 3: C=128 -> 64, 16x16 spatial
    kan_expand_nchw<<<nb(2097152, T), T, 0, stream>>>(P2, E3, 16 * 16, 2097152);
    pack_conv_w<<<nb(10368 * 64, T), T, 0, stream>>>(wb3, ws3, W3p, 128, 64, 10368);
    kan_conv_wmma<<<dim3(16384 / 128, 1), T, 0, stream>>>(E3, W3p, C3, 1152, 16, 16, 64, 16384, 10368);
    maxpool2x2<<<nb(262144, T), T, 0, stream>>>(C3, P3, 16, 16, 262144);

    // ---- KAN linear head: [64,4096] -> [64,100]
    kan_expand_rowmajor<<<nb(262144, T), T, 0, stream>>>(P3, EL, 262144);
    pack_linear_w<<<nb(4128768, T), T, 0, stream>>>(lb, lc, WLp, 4128768);
    zero_f32<<<nb(6400, T), T, 0, stream>>>(out, 6400);
    kan_linear_wmma<<<dim3(4, 7, 32), 32, 0, stream>>>(EL, WLp, out);
}
